// AdaptiveMemProcess_66941360275680
// MI455X (gfx1250) — matrix operand order repaired and checked by reading, not compiled, not run on hardware
//
#include <hip/hip_runtime.h>

typedef __attribute__((ext_vector_type(16))) __bf16 v16bf;
typedef __attribute__((ext_vector_type(2)))  __bf16 bf16x2;
typedef __attribute__((ext_vector_type(2)))  float  f32x2;
typedef __attribute__((ext_vector_type(8)))  float  v8f;

union FragU {
  v16bf v;
  unsigned short h[16];
  unsigned d[8];
  uint4 q[2];
};

// native f32 -> bf16 (hardware cvt)
__device__ __forceinline__ unsigned short f2bf(float f) {
  union { __bf16 b; unsigned short u; } cv;
  cv.b = (__bf16)f;
  return cv.u;
}

// packed pair convert -> should map to v_cvt_pk_bf16_f32
__device__ __forceinline__ unsigned f2bf2(float a, float b) {
  union { bf16x2 v; unsigned u; } cv;
  f32x2 t = {a, b};
  cv.v = __builtin_convertvector(t, bf16x2);
  return cv.u;
}

__device__ __forceinline__ v8f v8zero() {
  v8f z = {0.f,0.f,0.f,0.f,0.f,0.f,0.f,0.f};
  return z;
}

__device__ __forceinline__ v8f wmma_bf16(v16bf a, v16bf b, v8f c) {
  // D = A(16x32 bf16) x B(32x16 bf16) + C(f32)
  return __builtin_amdgcn_wmma_f32_16x16x32_bf16(false, a, false, b, (short)0, c, false, false);
}

// A-operand 16x32 tile: rows r0..r0+15, k0..k0+31, bf16 row-major, leading dim ld (elems).
// ISA layout: lanes 0-15 hold K[0..7],K[16..23]; lanes 16-31 hold K[8..15],K[24..31].
__device__ __forceinline__ v16bf ld_fragA(const unsigned short* p0, int ld, int r0, int k0) {
  int lane = threadIdx.x & 31;
  const unsigned short* p = p0 + (size_t)(r0 + (lane & 15)) * ld + k0 + ((lane >> 4) << 3);
  FragU f;
  f.q[0] = *(const uint4*)(p);
  f.q[1] = *(const uint4*)(p + 16);
  return f.v;
}

// B-operand 32x16 tile given as B^T rows (n0..n0+15), k0..k0+31, bf16 row-major.
// ISA layout: lanes 0-15 hold K[0..7],K[16..23]; lanes 16-31 hold K[8..15],K[24..31].
__device__ __forceinline__ v16bf ld_fragB(const unsigned short* p0, int ld, int n0, int k0) {
  int lane = threadIdx.x & 31;
  const unsigned short* p = p0 + (size_t)(n0 + (lane & 15)) * ld + k0 + ((lane >> 4) << 3);
  FragU f;
  f.q[0] = *(const uint4*)(p);
  f.q[1] = *(const uint4*)(p + 16);
  return f.v;
}

__device__ __forceinline__ v16bf ld_fragA_f32(const float* p0, int ld, int r0, int k0) {
  int lane = threadIdx.x & 31;
  const float* p = p0 + (size_t)(r0 + (lane & 15)) * ld + k0 + ((lane >> 4) << 3);
  FragU f;
#pragma unroll
  for (int i = 0; i < 4; ++i) f.d[i] = f2bf2(p[2 * i], p[2 * i + 1]);
#pragma unroll
  for (int i = 0; i < 4; ++i) f.d[4 + i] = f2bf2(p[16 + 2 * i], p[16 + 2 * i + 1]);
  return f.v;
}

__device__ __forceinline__ v16bf ld_fragB_f32(const float* p0, int ld, int n0, int k0) {
  int lane = threadIdx.x & 31;
  const float* p = p0 + (size_t)(n0 + (lane & 15)) * ld + k0 + ((lane >> 4) << 3);
  FragU f;
#pragma unroll
  for (int i = 0; i < 8; ++i) f.d[i] = f2bf2(p[2 * i + ((i >> 2) << 3)], p[2 * i + ((i >> 2) << 3) + 1]);
  return f.v;
}

__device__ __forceinline__ float sig_(float x) { return 1.f / (1.f + __expf(-x)); }
__device__ __forceinline__ float tanh_(float x) {
  x = fminf(fmaxf(x, -15.f), 15.f);
  float e = __expf(2.f * x);
  return (e - 1.f) / (e + 1.f);
}

// ---------------- prep kernels ----------------

__global__ void k_cvt2(const float* a, const float* b,
                       unsigned short* oa, unsigned short* ob, int n) {
  int i = blockIdx.x * blockDim.x + threadIdx.x;
  if (i < n) { oa[i] = f2bf(a[i]); ob[i] = f2bf(b[i]); }
}

// keys -> bf16 + per-row L2 norm. block=256 (8 waves), 1 wave per row.
__global__ void k_keys(const float* keys, unsigned short* kb, float* kn) {
  int row = blockIdx.x * 8 + (threadIdx.x >> 5);
  int lane = threadIdx.x & 31;
  const float* p = keys + (size_t)row * 256;
  float s = 0.f;
  for (int i = lane; i < 256; i += 32) {
    float v = p[i];
    kb[(size_t)row * 256 + i] = f2bf(v);
    s += v * v;
  }
#pragma unroll
  for (int m = 16; m >= 1; m >>= 1) s += __shfl_xor(s, m, 32);
  if (lane == 0) kn[row] = sqrtf(s);
}

// values (32768x256 f32) -> valuesT (256x32768 bf16), tiled transpose.
__global__ void k_valsT(const float* vals, unsigned short* vT) {
  __shared__ float tile[32][33];
  int s0 = blockIdx.x * 32, d0 = blockIdx.y * 32;
  int tx = threadIdx.x & 31, ty = threadIdx.x >> 5;  // 256 threads
#pragma unroll
  for (int i = 0; i < 4; ++i) {
    int r = ty + i * 8;
    tile[r][tx] = vals[(size_t)(s0 + r) * 256 + d0 + tx];
  }
  __syncthreads();
#pragma unroll
  for (int i = 0; i < 4; ++i) {
    int r = ty + i * 8;
    vT[(size_t)(d0 + r) * 32768 + s0 + tx] = f2bf(tile[tx][r]);
  }
}

// ---------------- embed: emb_bf16 = bf16(inp @ Wemb^T + bemb) ----------------
// grid (128,16), block 32 (one wave per 16x16 tile)
__global__ void k_embed(const float* inp, const float* Wemb, const float* bemb,
                        unsigned short* embbf) {
  int m0 = blockIdx.x * 16, n0 = blockIdx.y * 16;
  v8f acc = v8zero();
#pragma unroll
  for (int kc = 0; kc < 8; ++kc) {
    v16bf a = ld_fragA_f32(inp, 256, m0, kc * 32);
    v16bf b = ld_fragB_f32(Wemb, 256, n0, kc * 32);
    acc = wmma_bf16(a, b, acc);
  }
  int lane = threadIdx.x & 31, n = lane & 15, mh = (lane >> 4) * 8;
  float bias = bemb[n0 + n];
#pragma unroll
  for (int r = 0; r < 8; ++r)
    embbf[(size_t)(m0 + mh + r) * 256 + n0 + n] = f2bf(acc[r] + bias);
}

// ---------------- persistent LSTM: single workgroup, 16 waves ----------------
// wave w owns gate-dim slice [16w,16w+16) of all four gates -> in-register cell update.
__global__ void __launch_bounds__(512)
k_lstm(const unsigned short* embbf, const float* h0, const float* c0,
       const unsigned short* Wih, const unsigned short* Whh,
       const float* bih, const float* bhh, unsigned short* lstmbf) {
  __shared__ unsigned short hbf[16 * 256];
  int tid = threadIdx.x, w = tid >> 5, lane = tid & 31;
  int n = lane & 15, mh = (lane >> 4) * 8;
  int d = w * 16 + n;

  for (int e = tid; e < 4096; e += 512) hbf[e] = f2bf(h0[e]);

  float bb[4];
  v8f creg;
#pragma unroll
  for (int j = 0; j < 4; ++j) bb[j] = bih[j * 256 + d] + bhh[j * 256 + d];
#pragma unroll
  for (int r = 0; r < 8; ++r) creg[r] = c0[(mh + r) * 256 + d];
  __syncthreads();

#pragma unroll 1
  for (int t = 0; t < 128; ++t) {
    v8f acc[4];
#pragma unroll
    for (int j = 0; j < 4; ++j) acc[j] = v8zero();
#pragma unroll 1
    for (int kc = 0; kc < 8; ++kc) {
      v16bf ax = ld_fragA(embbf + (size_t)t * 256, 32768, 0, kc * 32);  // rows = batch
      v16bf ah = ld_fragA(hbf, 256, 0, kc * 32);
#pragma unroll
      for (int j = 0; j < 4; ++j) {
        int n0 = j * 256 + w * 16;
        v16bf b1 = ld_fragB(Wih, 256, n0, kc * 32);
        acc[j] = wmma_bf16(ax, b1, acc[j]);
        v16bf b2 = ld_fragB(Whh, 256, n0, kc * 32);
        acc[j] = wmma_bf16(ah, b2, acc[j]);
      }
    }
    __syncthreads();  // all reads of hbf for this step done
#pragma unroll
    for (int r = 0; r < 8; ++r) {
      float gi = acc[0][r] + bb[0];
      float gf = acc[1][r] + bb[1];
      float gg = acc[2][r] + bb[2];
      float go = acc[3][r] + bb[3];
      float cn = sig_(gf) * creg[r] + sig_(gi) * tanh_(gg);
      float h  = sig_(go) * tanh_(cn);
      creg[r] = cn;
      unsigned short hb = f2bf(h);
      hbf[(mh + r) * 256 + d] = hb;
      lstmbf[(size_t)((mh + r) * 128 + t) * 256 + d] = hb;
    }
    __syncthreads();  // h visible for next step
  }
}

// ---------------- out-proj: base = sigmoid(lstm @ Wout^T + bout) ----------------
__global__ void k_outproj(const unsigned short* lstmbf, const float* Wout,
                          const float* bout, float* out) {
  int m0 = blockIdx.x * 16, n0 = blockIdx.y * 16;
  v8f acc = v8zero();
#pragma unroll
  for (int kc = 0; kc < 8; ++kc) {
    v16bf a = ld_fragA(lstmbf, 256, m0, kc * 32);
    v16bf b = ld_fragB_f32(Wout, 256, n0, kc * 32);
    acc = wmma_bf16(a, b, acc);
  }
  int lane = threadIdx.x & 31, n = lane & 15, mh = (lane >> 4) * 8;
  float bias = bout[n0 + n];
#pragma unroll
  for (int r = 0; r < 8; ++r)
    out[(size_t)(m0 + mh + r) * 256 + n0 + n] = sig_(acc[r] + bias);
}

// ---------------- q = trg - base (bf16) + row norms ----------------
__global__ void k_qprep(const float* trg, const float* base,
                        unsigned short* qbf, float* qn) {
  int rowq = blockIdx.x;             // b*127 + t
  int b = rowq / 127, t = rowq % 127;
  int dd = threadIdx.x;
  size_t src = (size_t)(b * 128 + t) * 256 + dd;
  float v = trg[src] - base[src];
  qbf[(size_t)rowq * 256 + dd] = f2bf(v);
  float s = v * v;
#pragma unroll
  for (int m = 16; m >= 1; m >>= 1) s += __shfl_xor(s, m, 32);
  __shared__ float sred[8];
  if ((threadIdx.x & 31) == 0) sred[threadIdx.x >> 5] = s;
  __syncthreads();
  if (threadIdx.x == 0) {
    float tot = 0.f;
#pragma unroll
    for (int i = 0; i < 8; ++i) tot += sred[i];
    qn[rowq] = sqrtf(tot);
  }
}

// ---------------- attention over 32K memory slots ----------------
// grid (127 row-tiles, 8 S-slices), block 128 (4 waves). Each block handles
// 16 query rows x 4096 keys, emits partial contents (16x256 f32) + partial sums.
// sim in [-1,1] (cosine) -> streaming exp without running max.
__global__ void __launch_bounds__(128)
k_attn(const unsigned short* qbf, const float* qn,
       const unsigned short* keysbf, const float* kn,
       const unsigned short* vT, float* cpart, float* spart) {
  __shared__ unsigned short P[16 * 128];  // exp(sim) bf16, [query m][key-local]
  __shared__ float qn_s[16];
  __shared__ float sums[16];
  int tid = threadIdx.x, w = tid >> 5, lane = tid & 31;
  int n = lane & 15, mh = (lane >> 4) * 8;
  int row0 = blockIdx.x * 16;
  int sbeg = blockIdx.y * 4096;
  if (tid < 16) { qn_s[tid] = qn[row0 + tid]; sums[tid] = 0.f; }
  __syncthreads();

  v16bf qf[8];
#pragma unroll
  for (int kc = 0; kc < 8; ++kc) qf[kc] = ld_fragA(qbf, 256, row0, kc * 32);

  v8f cacc[4];
#pragma unroll
  for (int j = 0; j < 4; ++j) cacc[j] = v8zero();
  v8f srun = v8zero();

#pragma unroll 1
  for (int sb = sbeg; sb < sbeg + 4096; sb += 128) {
    if (sb + 128 < sbeg + 4096) {  // prefetch next chunk (keys + valuesT)
      __builtin_prefetch(keysbf + (size_t)(sb + 128 + w * 32 + lane) * 256, 0, 1);
      __builtin_prefetch(vT + (size_t)(w * 64 + lane) * 32768 + sb + 128, 0, 1);
    }
    // --- scores: wave w handles keys [sb+32w, sb+32w+32) as two 16-wide tiles
#pragma unroll
    for (int nt = 0; nt < 2; ++nt) {
      int sloc = w * 32 + nt * 16;
      int sg = sb + sloc;
      v8f a = v8zero();
#pragma unroll
      for (int kc = 0; kc < 8; ++kc) {
        v16bf bf = ld_fragB(keysbf, 256, sg, kc * 32);
        a = wmma_bf16(qf[kc], bf, a);
      }
      float knv = kn[sg + n];
#pragma unroll
      for (int r = 0; r < 8; ++r) {
        float rden = __builtin_amdgcn_rcpf(fmaxf(qn_s[mh + r] * knv, 1e-8f));
        float e = __expf(a[r] * rden);
        srun[r] += e;
        P[(mh + r) * 128 + sloc + n] = f2bf(e);
      }
    }
    __syncthreads();
    // --- contents: wave w owns output dims [64w, 64w+64)
#pragma unroll
    for (int kc = 0; kc < 4; ++kc) {
      v16bf pa = ld_fragA(P, 128, 0, kc * 32);
#pragma unroll
      for (int j = 0; j < 4; ++j) {
        int d0 = w * 64 + j * 16;
        v16bf bv = ld_fragB(vT, 32768, d0, sb + kc * 32);
        cacc[j] = wmma_bf16(pa, bv, cacc[j]);
      }
    }
    __syncthreads();
  }

  // reduce row sums: across the 16 lanes of each half-wave, then across waves
#pragma unroll
  for (int m = 8; m >= 1; m >>= 1) {
#pragma unroll
    for (int r = 0; r < 8; ++r) srun[r] += __shfl_xor(srun[r], m, 32);
  }
  if ((lane & 15) == 0) {
#pragma unroll
    for (int r = 0; r < 8; ++r) atomicAdd(&sums[mh + r], srun[r]);
  }
  __syncthreads();

  int ts = blockIdx.x * 8 + blockIdx.y;
#pragma unroll
  for (int j = 0; j < 4; ++j) {
    int d0 = w * 64 + j * 16;
#pragma unroll
    for (int r = 0; r < 8; ++r)
      cpart[(size_t)ts * 4096 + (mh + r) * 256 + d0 + n] = cacc[j][r];
  }
  if (tid < 16) spart[ts * 16 + tid] = sums[tid];
}

// ---------------- combine S-slice partials, out[b,t+1,:] += contents/sum ----------------
__global__ void k_combine(const float* cpart, const float* spart, float* out) {
  int rowq = blockIdx.x;                 // tile*16 + m  == b*127 + t
  int tile = rowq >> 4, m = rowq & 15;
  int dd = threadIdx.x;
  float tot = 0.f, stot = 0.f;
#pragma unroll
  for (int j = 0; j < 8; ++j) {
    int ts = tile * 8 + j;
    tot  += cpart[(size_t)ts * 4096 + m * 256 + dd];
    stot += spart[ts * 16 + m];
  }
  int b = rowq / 127, t = rowq % 127;
  out[(size_t)(b * 128 + t + 1) * 256 + dd] += tot / stot;
}

// ---------------- launcher ----------------

extern "C" void kernel_launch(void* const* d_in, const int* in_sizes, int n_in,
                              void* d_out, int out_size, void* d_ws, size_t ws_size,
                              hipStream_t stream) {
  (void)in_sizes; (void)n_in; (void)out_size; (void)ws_size;
  const float* inp  = (const float*)d_in[0];
  const float* trg  = (const float*)d_in[1];
  const float* h0   = (const float*)d_in[2];
  const float* c0   = (const float*)d_in[3];
  const float* Wemb = (const float*)d_in[4];
  const float* bemb = (const float*)d_in[5];
  const float* Wih  = (const float*)d_in[6];
  const float* Whh  = (const float*)d_in[7];
  const float* bih  = (const float*)d_in[8];
  const float* bhh  = (const float*)d_in[9];
  const float* Wout = (const float*)d_in[10];
  const float* bout = (const float*)d_in[11];
  const float* keys = (const float*)d_in[12];
  const float* vals = (const float*)d_in[13];
  float* out = (float*)d_out;

  char* ws = (char*)d_ws;
  size_t off = 0;
  auto alloc = [&](size_t bytes) {
    void* p = ws + off;
    off += (bytes + 255) & ~(size_t)255;
    return p;
  };
  unsigned short* keysbf = (unsigned short*)alloc((size_t)32768 * 256 * 2);
  unsigned short* vT     = (unsigned short*)alloc((size_t)32768 * 256 * 2);
  float*          kn     = (float*)alloc((size_t)32768 * 4);
  unsigned short* Wihbf  = (unsigned short*)alloc((size_t)1024 * 256 * 2);
  unsigned short* Whhbf  = (unsigned short*)alloc((size_t)1024 * 256 * 2);
  unsigned short* embbf  = (unsigned short*)alloc((size_t)2048 * 256 * 2);
  unsigned short* lstmbf = (unsigned short*)alloc((size_t)2048 * 256 * 2);
  unsigned short* qbf    = (unsigned short*)alloc((size_t)2032 * 256 * 2);
  float*          qn     = (float*)alloc((size_t)2032 * 4);
  float*          cpart  = (float*)alloc((size_t)127 * 8 * 16 * 256 * 4);
  float*          spart  = (float*)alloc((size_t)127 * 8 * 16 * 4);

  k_cvt2<<<(262144 + 255) / 256, 256, 0, stream>>>(Wih, Whh, Wihbf, Whhbf, 262144);
  k_keys<<<4096, 256, 0, stream>>>(keys, keysbf, kn);
  { dim3 g(1024, 8); k_valsT<<<g, 256, 0, stream>>>(vals, vT); }
  { dim3 g(128, 16); k_embed<<<g, 32, 0, stream>>>(inp, Wemb, bemb, embbf); }
  k_lstm<<<1, 512, 0, stream>>>(embbf, h0, c0, Wihbf, Whhbf, bih, bhh, lstmbf);
  { dim3 g(128, 16); k_outproj<<<g, 32, 0, stream>>>(lstmbf, Wout, bout, out); }
  k_qprep<<<2032, 256, 0, stream>>>(trg, out, qbf, qn);
  { dim3 g(127, 8); k_attn<<<g, 128, 0, stream>>>(qbf, qn, keysbf, kn, vT, cpart, spart); }
  k_combine<<<2032, 256, 0, stream>>>(cpart, spart, out);
}
